// FusionModule_39084202393928
// MI455X (gfx1250) — matrix operand order repaired and checked by reading, compiled, not run on hardware
//
#include <hip/hip_runtime.h>
#include <math.h>

typedef unsigned short u16;

// ---------------------------------------------------------------------------
// CDNA5 (gfx1250) WMMA types & helpers
// ---------------------------------------------------------------------------
typedef __attribute__((ext_vector_type(16))) __bf16 v16bf;
typedef __attribute__((ext_vector_type(8)))  float  v8f;

union FragU { uint4 q[2]; u16 s[16]; v16bf v; };

__device__ __forceinline__ u16 f2bf_bits(float f) {
    union { float f; unsigned u; } a; a.f = f;
    unsigned u = a.u;
    return (u16)((u + 0x7FFFu + ((u >> 16) & 1u)) >> 16);   // RNE
}

__device__ __forceinline__ v8f wmma_bf16(v16bf a, v16bf b, v8f c) {
    return __builtin_amdgcn_wmma_f32_16x16x32_bf16(false, a, false, b,
                                                   (short)0, c, false, false);
}

// A-matrix 16x32 bf16 fragment: from a K-contiguous row.
// lanes 0-15: elems0-15 -> K+0..15
// lanes16-31: elems0-15 -> K+16..31
__device__ __forceinline__ v16bf load_afrag(const u16* p, int laneHi) {
    FragU f;
    f.q[0] = *(const uint4*)(p + (laneHi << 4));
    f.q[1] = *(const uint4*)(p + 8 + (laneHi << 4));
    return f.v;
}
// B-matrix 32x16 bf16 fragment: per lane one column, contiguous 16-K half.
// lanes 0-15: K+0..15 ; lanes 16-31: K+16..31
__device__ __forceinline__ v16bf load_bfrag(const u16* p, int laneHi) {
    FragU f;
    const uint4* q = (const uint4*)(p + (laneHi << 4));
    f.q[0] = q[0]; f.q[1] = q[1];
    return f.v;
}

// ---------------------------------------------------------------------------
// Prep: fp32 -> bf16 copy
// ---------------------------------------------------------------------------
__global__ void pack_bf16_kernel(const float* __restrict__ src,
                                 u16* __restrict__ dst, long total) {
    long i = (long)blockIdx.x * blockDim.x + threadIdx.x;
    if (i < total) dst[i] = f2bf_bits(src[i]);
}

// Fold w_q for edge_cat=[e;e]: Wq_eff[o][c] = w_q[o][c] + w_q[o][c+128], bf16
__global__ void build_wq_eff_kernel(const float* __restrict__ wq,
                                    u16* __restrict__ wq_eff) {
    int i = blockIdx.x * blockDim.x + threadIdx.x;
    if (i >= 256 * 128) return;
    int o = i >> 7, c = i & 127;
    wq_eff[i] = f2bf_bits(wq[o * 256 + c] + wq[o * 256 + 128 + c]);
}

// ---------------------------------------------------------------------------
// Bilinear 2x upsample (align_corners=False) 32x32 -> 64x64, output bf16
// transposed: dst[b][n][c]
// ---------------------------------------------------------------------------
__global__ void upsample2x_bf16_kernel(const float* __restrict__ src,
                                       u16* __restrict__ dst,
                                       int Cshift, long total) {
    long i = (long)blockIdx.x * blockDim.x + threadIdx.x;
    if (i >= total) return;
    const int C = 1 << Cshift;
    int c = (int)(i & (C - 1));
    long nb = i >> Cshift;
    int n = (int)(nb & 4095);
    int b = (int)(nb >> 12);
    int y = n >> 6, x = n & 63;
    float sy = 0.5f * (float)y - 0.25f;
    float sx = 0.5f * (float)x - 0.25f;
    int y0 = (int)floorf(sy); float fy = sy - (float)y0;
    int x0 = (int)floorf(sx); float fx = sx - (float)x0;
    int y0c = min(max(y0, 0), 31), y1c = min(max(y0 + 1, 0), 31);
    int x0c = min(max(x0, 0), 31), x1c = min(max(x0 + 1, 0), 31);
    const float* s = src + ((long)b * C + c) * 1024;
    float v00 = s[y0c * 32 + x0c], v01 = s[y0c * 32 + x1c];
    float v10 = s[y1c * 32 + x0c], v11 = s[y1c * 32 + x1c];
    float v = (1.f - fy) * ((1.f - fx) * v00 + fx * v01)
            +        fy  * ((1.f - fx) * v10 + fx * v11);
    dst[i] = f2bf_bits(v);
}

// ---------------------------------------------------------------------------
// WMMA GEMM, compile-time K (software-pipelined, fully unrolled).
//   Y[b,co,n] = sum_k W[co,k] * X(b,k,n) + bias[co]
// Wb: bf16 [Cout][K0+K1].  X = concat_k of X0t [B][N][K0], X1t [B][N][K1].
// One wave -> 16co x 64n tile. Outputs (null = skip):
//   out_t: bf16 [B][N][Cout]  out_tf: f32 [B][N][Cout]
//   out_cf: f32 [B][Cout][N]  out_cb: bf16 [B][Cout][N]
// ---------------------------------------------------------------------------
template <int K0, int K1>
__global__ void __launch_bounds__(32)
gemm_wmma_kernel(const u16* __restrict__ Wb, const float* __restrict__ bias,
                 const u16* __restrict__ X0, const u16* __restrict__ X1,
                 u16* __restrict__ out_t, float* __restrict__ out_tf,
                 float* __restrict__ out_cf, u16* __restrict__ out_cb,
                 int Cout, int N) {
    constexpr int Ktot = K0 + K1;
    constexpr int NCH  = Ktot / 32;

    const int lane   = threadIdx.x & 31;
    const int laneHi = lane >> 4;
    const int lmod   = lane & 15;
    const int n0  = blockIdx.x * 64;
    const int co0 = blockIdx.y * 16;
    const int b   = blockIdx.z;

    const u16* wrow = Wb + (size_t)(co0 + lmod) * Ktot;
    const u16* x0r  = X0 + ((size_t)b * N + n0 + lmod) * K0;
    const u16* x1r  = X1 + ((size_t)b * N + n0 + lmod) * K1;

    auto ldA = [&](int ch) { return load_afrag(wrow + ch * 32, laneHi); };
    auto ldB = [&](int ch, int t) {
        if (ch * 32 < K0)
            return load_bfrag(x0r + (size_t)(t * 16) * K0 + ch * 32, laneHi);
        else
            return load_bfrag(x1r + (size_t)(t * 16) * K1 + (ch * 32 - K0), laneHi);
    };

    v8f acc[4] = {};
    v16bf a_c = ldA(0);
    v16bf b_c0 = ldB(0, 0), b_c1 = ldB(0, 1), b_c2 = ldB(0, 2), b_c3 = ldB(0, 3);
    __builtin_prefetch(wrow, 0, 1);                       // global_prefetch_b8

#pragma unroll
    for (int ch = 0; ch < NCH; ++ch) {
        v16bf a_n = a_c, b_n0 = b_c0, b_n1 = b_c1, b_n2 = b_c2, b_n3 = b_c3;
        if (ch + 1 < NCH) {           // compile-time after unroll
            a_n  = ldA(ch + 1);
            b_n0 = ldB(ch + 1, 0); b_n1 = ldB(ch + 1, 1);
            b_n2 = ldB(ch + 1, 2); b_n3 = ldB(ch + 1, 3);
        }
        acc[0] = wmma_bf16(a_c, b_c0, acc[0]);
        acc[1] = wmma_bf16(a_c, b_c1, acc[1]);
        acc[2] = wmma_bf16(a_c, b_c2, acc[2]);
        acc[3] = wmma_bf16(a_c, b_c3, acc[3]);
        a_c = a_n; b_c0 = b_n0; b_c1 = b_n1; b_c2 = b_n2; b_c3 = b_n3;
    }

    float bs[8];
#pragma unroll
    for (int r = 0; r < 8; ++r) bs[r] = bias[co0 + r + (laneHi << 3)];

#pragma unroll
    for (int t = 0; t < 4; ++t) {
        int n = n0 + t * 16 + lmod;
        float v[8];
#pragma unroll
        for (int r = 0; r < 8; ++r) v[r] = acc[t][r] + bs[r];

        if (out_t) {
            u16* dst = out_t + ((size_t)b * N + n) * Cout + co0 + (laneHi << 3);
#pragma unroll
            for (int r = 0; r < 8; r += 2) {
                unsigned p = (unsigned)f2bf_bits(v[r]) |
                             ((unsigned)f2bf_bits(v[r + 1]) << 16);
                *(unsigned*)(dst + r) = p;
            }
        }
        if (out_tf) {
            float* dst = out_tf + ((size_t)b * N + n) * Cout + co0 + (laneHi << 3);
#pragma unroll
            for (int r = 0; r < 8; r += 2) {
                float2 p; p.x = v[r]; p.y = v[r + 1];
                *(float2*)(dst + r) = p;
            }
        }
        if (out_cf) {
#pragma unroll
            for (int r = 0; r < 8; ++r)
                out_cf[((size_t)b * Cout + co0 + r + (laneHi << 3)) * N + n] = v[r];
        }
        if (out_cb) {
#pragma unroll
            for (int r = 0; r < 8; ++r)
                out_cb[((size_t)b * Cout + co0 + r + (laneHi << 3)) * N + n] = f2bf_bits(v[r]);
        }
    }
}

// ---------------------------------------------------------------------------
// similarity + density softmax.
// ---------------------------------------------------------------------------
__global__ void sim_density_kernel(const float* __restrict__ edge_al_t,
                                   const float* __restrict__ sem,
                                   u16* __restrict__ edge_t,
                                   u16* __restrict__ sem_t) {
    __shared__ float l2e_s[4096];
    __shared__ float sim_s[4096];
    __shared__ float red[1024];
    const int b = blockIdx.x, t = threadIdx.x;
    const float* E = edge_al_t + (size_t)b * 4096 * 128;
    const float* S = sem       + (size_t)b * 128 * 4096;

    for (int p = 0; p < 4; ++p) {
        int n = t + p * 1024;
        float se = 0.f, ss = 0.f, dot = 0.f;
        for (int c = 0; c < 128; ++c) {
            float e = E[(size_t)n * 128 + c];
            float s = S[(size_t)c * 4096 + n];
            se += e * e; ss += s * s; dot += e * s;
        }
        float l2e = sqrtf(se), l2s = sqrtf(ss);
        float sim = dot / ((l2e + 1e-6f) * (l2s + 1e-6f));
        l2e_s[n] = l2e;
        sim_s[n] = (sim + 1.0f) * 0.5f;
    }
    __syncthreads();

    float m = fmaxf(fmaxf(l2e_s[t], l2e_s[t + 1024]),
                    fmaxf(l2e_s[t + 2048], l2e_s[t + 3072]));
    red[t] = m; __syncthreads();
    for (int s = 512; s > 0; s >>= 1) {
        if (t < s) red[t] = fmaxf(red[t], red[t + s]);
        __syncthreads();
    }
    const float mx = red[0]; __syncthreads();

    float es = 0.f;
    for (int p = 0; p < 4; ++p) es += expf(l2e_s[t + p * 1024] - mx);
    red[t] = es; __syncthreads();
    for (int s = 512; s > 0; s >>= 1) {
        if (t < s) red[t] += red[t + s];
        __syncthreads();
    }
    const float inv_sum = 4096.0f / red[0];
    __syncthreads();

    u16* ET = edge_t + (size_t)b * 4096 * 128;
    u16* ST = sem_t  + (size_t)b * 4096 * 128;
    for (int p = 0; p < 4; ++p) {
        int n = t + p * 1024;
        float sim = sim_s[n];
        float den = expf(l2e_s[n] - mx) * inv_sum;
        for (int c = 0; c < 128; ++c) {
            ET[(size_t)n * 128 + c] = f2bf_bits(E[(size_t)n * 128 + c] * sim);
            ST[(size_t)n * 128 + c] = f2bf_bits(S[(size_t)c * 4096 + n] * den);
        }
    }
}

// ---------------------------------------------------------------------------
// Windowed attention (11x11). Grid (4 qy-tiles, 64 qx, B), 1 wave/block.
// Key slots: 11 rows x 32 cols, column window starts at qy0-8 (aligned).
// Score-phase K loads are BRANCHLESS: out-of-image key addresses are clamped
// into the image (their scores are overwritten with -1e30 in the softmax
// masking pass, so finite garbage is harmless). This keeps EXEC uniform and
// lets the scheduler clause all 16 loads of a key block ahead of the WMMAs.
// ---------------------------------------------------------------------------
__global__ void __launch_bounds__(32)
attn_wmma_kernel(const u16* __restrict__ qt, const u16* __restrict__ kt,
                 const u16* __restrict__ semb, const u16* __restrict__ fusedb,
                 u16* __restrict__ out_t) {
    __shared__ float Ss[16 * 352];
    __shared__ u16   Pb[16 * 352];

    const int lane   = threadIdx.x & 31;
    const int laneHi = lane >> 4;
    const int lmod   = lane & 15;
    const int qy0 = blockIdx.x * 16;
    const int qx  = blockIdx.y;
    const int b   = blockIdx.z;
    const int nq0 = qx * 64 + qy0;

    const u16* Qb = qt + (size_t)b * 4096 * 256;
    const u16* Kb = kt + (size_t)b * 4096 * 256;

    // stage all 8 Q A-fragments (K = 256) in registers, reuse for 22 blocks
    v16bf qf[8];
    {
        const u16* qrow = Qb + (size_t)(nq0 + lmod) * 256;
#pragma unroll
        for (int c8 = 0; c8 < 8; ++c8)
            qf[c8] = load_afrag(qrow + c8 * 32, laneHi);
    }

    // ---- scores: S[16 x 352], branchless clamped loads ----
    for (int blk = 0; blk < 22; ++blk) {
        int slot = blk * 16 + lmod;
        int kxi = slot >> 5, kyi = slot & 31;
        int kxr = min(max(qx - 5 + kxi, 0), 63);
        int kyc = min(max(qy0 - 8 + kyi, 0), 63);
        const u16* krow = Kb + (size_t)(kxr * 64 + kyc) * 256;

        v8f acc = {};
#pragma unroll
        for (int c8 = 0; c8 < 8; ++c8)
            acc = wmma_bf16(qf[c8], load_bfrag(krow + c8 * 32, laneHi), acc);
#pragma unroll
        for (int r = 0; r < 8; ++r)
            Ss[(r + (laneHi << 3)) * 352 + blk * 16 + lmod] = acc[r];
    }
    __syncthreads();

    // ---- masked softmax, in LDS. scale = 1/sqrt(256) = 0.0625 ----
    {
        const int qi  = lmod;              // row
        const int s0  = laneHi * 176;      // slot half
        const int qyq = qy0 + qi;
        float* row = Ss + qi * 352;

        float mx = -1e30f;
        for (int s = s0; s < s0 + 176; ++s) {
            int kxi = s >> 5, kyi = s & 31;
            int kxr = qx - 5 + kxi, kyc = qy0 - 8 + kyi;
            bool valid = (kxr >= 0) && (kxr < 64) && (kyc >= 0) && (kyc < 64) &&
                         (abs(kyc - qyq) <= 5);
            float v = valid ? row[s] * 0.0625f : -1e30f;
            row[s] = v;
            mx = fmaxf(mx, v);
        }
        mx = fmaxf(mx, __shfl_xor(mx, 16, 32));

        float sum = 0.f;
        for (int s = s0; s < s0 + 176; ++s) {
            float p = expf(row[s] - mx);   // masked -> exactly 0
            row[s] = p;
            sum += p;
        }
        sum += __shfl_xor(sum, 16, 32);
        float inv = 1.0f / sum;
        for (int s = s0; s < s0 + 176; ++s)
            Pb[qi * 352 + s] = f2bf_bits(row[s] * inv);
    }
    __syncthreads();

    // ---- O = P V (V runs must stay position-correct: keep edge fallback) ----
#pragma unroll 1
    for (int cb2 = 0; cb2 < 256; cb2 += 16) {
        int c = cb2 + lmod;
        const u16* vbase = (c < 128)
            ? semb   + ((size_t)b * 128 + c) * 4096
            : fusedb + ((size_t)b * 128 + (c - 128)) * 4096;

        v8f acc = {};
#pragma unroll
        for (int mk = 0; mk < 352; mk += 32) {
            v16bf a = load_afrag(Pb + lmod * 352 + mk, laneHi);
            int kxr = qx - 5 + (mk >> 5);
            int ky0 = qy0 - 8 + (laneHi << 4);
            FragU f;
            if (kxr >= 0 && kxr < 64 && ky0 >= 0 && ky0 + 15 < 64) {
                const uint4* qp = (const uint4*)(vbase + kxr * 64 + ky0);
                f.q[0] = qp[0]; f.q[1] = qp[1];
            } else {
#pragma unroll
                for (int j = 0; j < 16; ++j) {
                    int ky = ky0 + j;
                    f.s[j] = (kxr >= 0 && kxr < 64 && ky >= 0 && ky < 64)
                           ? vbase[kxr * 64 + ky] : (u16)0;
                }
            }
            acc = wmma_bf16(a, f.v, acc);
        }
#pragma unroll
        for (int r = 0; r < 8; ++r) {
            int qi = r + (laneHi << 3);
            out_t[((size_t)b * 4096 + nq0 + qi) * 256 + c] = f2bf_bits(acc[r]);
        }
    }
}

// ---------------------------------------------------------------------------
// Launch
// ---------------------------------------------------------------------------
extern "C" void kernel_launch(void* const* d_in, const int* in_sizes, int n_in,
                              void* d_out, int out_size, void* d_ws, size_t ws_size,
                              hipStream_t stream) {
    (void)in_sizes; (void)n_in; (void)out_size; (void)ws_size;
    const float* edge_features  = (const float*)d_in[0];
    const float* sem            = (const float*)d_in[1];
    const float* fused_features = (const float*)d_in[2];
    const float* w_align        = (const float*)d_in[3];
    const float* b_align        = (const float*)d_in[4];
    const float* w_fused_align  = (const float*)d_in[5];
    const float* b_fused_align  = (const float*)d_in[6];
    const float* w_q            = (const float*)d_in[7];
    const float* b_q            = (const float*)d_in[8];
    const float* w_k            = (const float*)d_in[9];
    const float* b_k            = (const float*)d_in[10];
    const float* w_fusion       = (const float*)d_in[11];
    const float* b_fusion       = (const float*)d_in[12];
    float* out = (float*)d_out;

    const long N = 4096, B = 4;
    char* wsp = (char*)d_ws;
    auto alloc = [&](size_t bytes) -> char* {
        char* p = wsp; wsp += (bytes + 255) & ~(size_t)255; return p;
    };
    u16* edge_up    = (u16*)alloc(B * N * 64 * 2);
    u16* fused_up   = (u16*)alloc(B * N * 128 * 2);
    u16* semb       = (u16*)alloc(B * 128 * N * 2);
    u16* wb_align   = (u16*)alloc(128 * 64 * 2);
    u16* wb_fused   = (u16*)alloc(128 * 128 * 2);
    u16* wb_qeff    = (u16*)alloc(256 * 128 * 2);
    u16* wb_k       = (u16*)alloc(256 * 256 * 2);
    u16* wb_fusion  = (u16*)alloc(128 * 256 * 2);
    float* edge_al  = (float*)alloc(B * N * 128 * 4);   // fp32 [B][N][128]
    u16* fused_al_t = (u16*)alloc(B * N * 128 * 2);     // bf16 [B][N][128]
    u16* fused_al_c = (u16*)alloc(B * 128 * N * 2);     // bf16 [B][128][N]
    u16* edge_t     = (u16*)alloc(B * N * 128 * 2);
    u16* sem_t      = (u16*)alloc(B * N * 128 * 2);
    u16* qt         = (u16*)alloc(B * N * 256 * 2);
    u16* kt         = (u16*)alloc(B * N * 256 * 2);
    u16* attn_t     = (u16*)alloc(B * N * 256 * 2);

    // prep
    {
        long tot = B * N * 64;
        upsample2x_bf16_kernel<<<(tot + 255) / 256, 256, 0, stream>>>(edge_features, edge_up, 6, tot);
        tot = B * N * 128;
        upsample2x_bf16_kernel<<<(tot + 255) / 256, 256, 0, stream>>>(fused_features, fused_up, 7, tot);
        pack_bf16_kernel<<<(128 * 64 + 255) / 256, 256, 0, stream>>>(w_align, wb_align, 128 * 64);
        pack_bf16_kernel<<<(128 * 128 + 255) / 256, 256, 0, stream>>>(w_fused_align, wb_fused, 128 * 128);
        pack_bf16_kernel<<<(256 * 256 + 255) / 256, 256, 0, stream>>>(w_k, wb_k, 256 * 256);
        pack_bf16_kernel<<<(128 * 256 + 255) / 256, 256, 0, stream>>>(w_fusion, wb_fusion, 128 * 256);
        pack_bf16_kernel<<<(B * 128 * N + 255) / 256, 256, 0, stream>>>(sem, semb, B * 128 * N);
        build_wq_eff_kernel<<<128, 256, 0, stream>>>(w_q, wb_qeff);
    }

    // alignment convs
    dim3 g128(64, 8, 4);
    gemm_wmma_kernel<64, 0><<<g128, 32, 0, stream>>>(wb_align, b_align,
        edge_up, edge_up, nullptr, edge_al, nullptr, nullptr, 128, (int)N);
    gemm_wmma_kernel<128, 0><<<g128, 32, 0, stream>>>(wb_fused, b_fused_align,
        fused_up, fused_up, fused_al_t, nullptr, nullptr, fused_al_c, 128, (int)N);

    // similarity + density
    sim_density_kernel<<<4, 1024, 0, stream>>>(edge_al, sem, edge_t, sem_t);

    // Q/K projections
    dim3 g256(64, 16, 4);
    gemm_wmma_kernel<128, 0><<<g256, 32, 0, stream>>>(wb_qeff, b_q,
        edge_t, edge_t, qt, nullptr, nullptr, nullptr, 256, (int)N);
    gemm_wmma_kernel<128, 128><<<g256, 32, 0, stream>>>(wb_k, b_k,
        sem_t, fused_al_t, kt, nullptr, nullptr, nullptr, 256, (int)N);

    // windowed attention
    dim3 ga(4, 64, 4);
    attn_wmma_kernel<<<ga, 32, 0, stream>>>(qt, kt, semb, fused_al_c, attn_t);

    // fusion conv -> d_out (fp32 [B][128][N])
    gemm_wmma_kernel<256, 0><<<g128, 32, 0, stream>>>(wb_fusion, b_fusion,
        attn_t, attn_t, nullptr, nullptr, out, nullptr, 128, (int)N);
}
